// MultiHeadAttention_16080357556499
// MI455X (gfx1250) — matrix operand order repaired and checked by reading, not compiled, not run on hardware
//
#include <hip/hip_runtime.h>
#include <hip/hip_bf16.h>

// ---------------------------------------------------------------------------
// MHA forward for MI455X (gfx1250, wave32, WMMA).
// [qkv proj f32->f16 GEMM, 128x64 block tile] ->
// [flash attention, async K-tile staging, f16 WMMA, fp32 online softmax] ->
// [output proj GEMM, async A-tile staging, fp32 out].
// All matrix math: v_wmma_f32_16x16x32_f16 (fp32 accumulate).
// ---------------------------------------------------------------------------

typedef __attribute__((ext_vector_type(16))) _Float16 v16h;
typedef __attribute__((ext_vector_type(8)))  _Float16 v8h;
typedef __attribute__((ext_vector_type(8)))  float    v8f;

#define DEV static __device__ __forceinline__

constexpr int Bn = 4, Sn = 2048, Hn = 1024, NHn = 16, HDn = 64;
constexpr size_t QKV_ELEMS = (size_t)Bn * NHn * Sn * HDn;   // 8,388,608 halfs (16MB)

union FragU { v16h v; v8h h[2]; };

// A-matrix fragment (16 rows x 32 k), source row-major [row][k], ldk halfs.
// Lane L: m = L&15, hi = L>>4; lanes 0-15 hold k {0..7,16..23},
// lanes 16-31 hold k {8..15,24..31}.
DEV v16h load_a_frag(const _Float16* base, int row, int ldk, int kbase, int hi) {
  const _Float16* p = base + (size_t)row * ldk + kbase + hi * 8;
  FragU u;
  u.h[0] = *(const v8h*)(p);
  u.h[1] = *(const v8h*)(p + 16);
  return u.v;
}

// B-matrix fragment (32 k x 16 cols), source stored Bt[n][k] (column of B
// contiguous in k). Lane L: n = L&15; lanes 0-15 hold k {0..7,16..23},
// lanes 16-31 hold k {8..15,24..31}.
DEV v16h load_b_frag(const _Float16* base, int n, int ldk, int kbase, int hi) {
  const _Float16* p = base + (size_t)n * ldk + kbase + hi * 8;
  FragU u;
  u.h[0] = *(const v8h*)(p);
  u.h[1] = *(const v8h*)(p + 16);
  return u.v;
}

DEV v8f wmma_f16(v16h a, v16h b, v8f c) {
  return __builtin_amdgcn_wmma_f32_16x16x32_f16(false, a, false, b, (short)0, c,
                                                false, false);
}

// Async raw copy of 16 bytes global -> LDS (gfx1250 GLOBAL_LOAD_ASYNC_TO_LDS).
DEV void async_copy_b128(void* ldsDst, const void* gsrc) {
  unsigned ldsOff = (unsigned)(size_t)ldsDst;   // low 32 bits = LDS offset
  asm volatile("global_load_async_to_lds_b128 %0, %1, off"
               :: "v"(ldsOff), "v"(gsrc) : "memory");
}
DEV void async_wait0() {
  asm volatile("s_wait_asynccnt 0" ::: "memory");
}

// ---------------------------------------------------------------------------
// Kernel 1: fused QKV projection. Block = 8 waves, tile 128(M) x 64(N),
// K-step 32 through LDS; each wave: 1 A-frag reused over 4 B-frags = 4 WMMAs.
// Output scattered to (B, NH, S, HD) f16.
// ---------------------------------------------------------------------------
__global__ __launch_bounds__(256)
void qkv_proj_kernel(const float* __restrict__ x,
                     const float* __restrict__ Wq, const float* __restrict__ bq,
                     const float* __restrict__ Wk, const float* __restrict__ bk,
                     const float* __restrict__ Wv, const float* __restrict__ bv,
                     _Float16* __restrict__ qws, _Float16* __restrict__ kws,
                     _Float16* __restrict__ vws)
{
  __shared__ __align__(16) _Float16 Als[128 * 32];   // 8KB
  __shared__ __align__(16) _Float16 Bls[64 * 32];    // 4KB

  const int t   = threadIdx.x;
  const int m0  = blockIdx.x * 128;
  const int nG0 = blockIdx.y * 64;
  const int mid = nG0 >> 10;            // 0=Q 1=K 2=V
  const int n0  = nG0 & (Hn - 1);
  const float* W    = (mid == 0) ? Wq : (mid == 1) ? Wk : Wv;
  const float* bias = (mid == 0) ? bq : (mid == 1) ? bk : bv;
  _Float16*    ows  = (mid == 0) ? qws : (mid == 1) ? kws : vws;

  const int lane = t & 31, w = t >> 5;
  const int hi = lane >> 4, mloc = lane & 15;

  const v8f vzero = {0.f, 0.f, 0.f, 0.f, 0.f, 0.f, 0.f, 0.f};
  v8f acc[4] = {vzero, vzero, vzero, vzero};

  for (int k0 = 0; k0 < Hn; k0 += 32) {
    // speculative prefetch of next K-step tiles (global_prefetch_b8)
    if (k0 + 32 < Hn) {
      if (t < 128)
        __builtin_prefetch(x + (size_t)(m0 + t) * Hn + (k0 + 32), 0, 3);
      else
        __builtin_prefetch(W + (size_t)(n0 + (t & 63)) * Hn + (k0 + 32), 0, 3);
    }
    // stage x tile 128x32 fp32 -> f16 (4 chunks of 4 per thread)
    #pragma unroll
    for (int i = 0; i < 4; ++i) {
      int c = t + i * 256;
      int row = c >> 3, kc = (c & 7) * 4;
      float4 f = *(const float4*)(x + (size_t)(m0 + row) * Hn + k0 + kc);
      _Float16* dst = &Als[row * 32 + kc];
      dst[0] = (_Float16)f.x; dst[1] = (_Float16)f.y;
      dst[2] = (_Float16)f.z; dst[3] = (_Float16)f.w;
    }
    // stage W tile 64(n) x 32(k) fp32 -> f16, stored [n][k]
    #pragma unroll
    for (int i = 0; i < 2; ++i) {
      int c = t + i * 256;
      int row = c >> 3, kc = (c & 7) * 4;
      float4 f = *(const float4*)(W + (size_t)(n0 + row) * Hn + k0 + kc);
      _Float16* dst = &Bls[row * 32 + kc];
      dst[0] = (_Float16)f.x; dst[1] = (_Float16)f.y;
      dst[2] = (_Float16)f.z; dst[3] = (_Float16)f.w;
    }
    __syncthreads();
    v16h a = load_a_frag(Als, w * 16 + mloc, 32, 0, hi);
    #pragma unroll
    for (int nb = 0; nb < 4; ++nb) {
      v16h b = load_b_frag(Bls, nb * 16 + mloc, 32, 0, hi);
      acc[nb] = wmma_f16(a, b, acc[nb]);
    }
    __syncthreads();
  }

  #pragma unroll
  for (int nb = 0; nb < 4; ++nb) {
    const int col  = n0 + nb * 16 + mloc;   // column inside this weight matrix
    const float bv_ = bias[col];
    const int head = col >> 6, d = col & 63;
    #pragma unroll
    for (int r = 0; r < 8; ++r) {
      int gm = m0 + w * 16 + r + hi * 8;
      int bb = gm >> 11;                    // / Sn
      int ss = gm & (Sn - 1);
      size_t dst = (((size_t)(bb * NHn + head)) * Sn + ss) * HDn + d;
      ows[dst] = (_Float16)(acc[nb][r] + bv_);
    }
  }
}

// ---------------------------------------------------------------------------
// Kernel 2: flash attention. Block = 128 query rows of one (batch, head);
// 8 waves x 16 query rows; loop over 32-key tiles. K tile staged with
// async-to-LDS b128; V transposed at LDS-store; online softmax in fp32;
// P reshaped C->A layout through per-wave LDS.
// ---------------------------------------------------------------------------
__global__ __launch_bounds__(256)
void flash_attn_kernel(const _Float16* __restrict__ qws,
                       const _Float16* __restrict__ kws,
                       const _Float16* __restrict__ vws,
                       const int* __restrict__ mask,
                       _Float16* __restrict__ cws)
{
  __shared__ __align__(16) _Float16 Kls[32 * 64];       // [key][d]
  __shared__ __align__(16) _Float16 Vtls[64 * 32];      // [d][key] (transposed)
  __shared__ __align__(16) _Float16 Pls[8][16 * 32];    // per-wave P tile [m][k]
  __shared__ float biasls[32];

  const int t = threadIdx.x;
  const int lane = t & 31, w = t >> 5;
  const int hi = lane >> 4, mloc = lane & 15;

  const int qt = blockIdx.x, head = blockIdx.y, bb = blockIdx.z;
  const size_t headOff = ((size_t)(bb * NHn + head)) * Sn * HDn;
  const _Float16* Qh = qws + headOff;
  const _Float16* Kh = kws + headOff;
  const _Float16* Vh = vws + headOff;

  const int q0 = qt * 128 + w * 16;

  // Q fragments (16 rows x 64 d) straight from global (b128 loads)
  v16h qf0 = load_a_frag(Qh, q0 + mloc, HDn, 0,  hi);
  v16h qf1 = load_a_frag(Qh, q0 + mloc, HDn, 32, hi);

  float mrun[8], lrun[8], alpha[8];
  const v8f vzero = {0.f, 0.f, 0.f, 0.f, 0.f, 0.f, 0.f, 0.f};
  v8f O0 = vzero, O1 = vzero, O2 = vzero, O3 = vzero;
  #pragma unroll
  for (int r = 0; r < 8; ++r) { mrun[r] = -__builtin_inff(); lrun[r] = 0.f; }

  for (int kb = 0; kb < Sn; kb += 32) {
    // ---- cooperative staging of 32-key K/V tile ----
    {
      int row = t >> 3, dc = (t & 7) * 8;   // 256 chunks of 8 halfs
      // K tile: raw f16 copy -> async DMA into LDS
      async_copy_b128(&Kls[row * 64 + dc],
                      Kh + (size_t)(kb + row) * HDn + dc);
      // V tile: needs transpose -> regular load + scattered LDS stores
      v8h vv = *(const v8h*)(Vh + (size_t)(kb + row) * HDn + dc);
      #pragma unroll
      for (int j = 0; j < 8; ++j) Vtls[(dc + j) * 32 + row] = vv[j];
    }
    if (t < 32)
      biasls[t] = (mask[bb * Sn + kb + t] == 0) ? -__builtin_inff() : 0.f;
    async_wait0();
    __syncthreads();

    // ---- scores S = Q . K^T (two 16-key subtiles, K-dim 64 = 2 WMMAs) ----
    v8f s0 = vzero, s1 = vzero;
    {
      v16h b00 = load_b_frag(Kls, mloc,      64, 0,  hi);
      v16h b01 = load_b_frag(Kls, mloc,      64, 32, hi);
      s0 = wmma_f16(qf0, b00, s0);
      s0 = wmma_f16(qf1, b01, s0);
      v16h b10 = load_b_frag(Kls, 16 + mloc, 64, 0,  hi);
      v16h b11 = load_b_frag(Kls, 16 + mloc, 64, 32, hi);
      s1 = wmma_f16(qf0, b10, s1);
      s1 = wmma_f16(qf1, b11, s1);
    }

    // ---- online softmax update (row = r + hi*8, stats per 16-lane half) ----
    const float bias0 = biasls[mloc], bias1 = biasls[16 + mloc];
    #pragma unroll
    for (int r = 0; r < 8; ++r) {
      float v0 = s0[r] * 0.125f + bias0;   // 1/sqrt(64)
      float v1 = s1[r] * 0.125f + bias1;
      float rm = fmaxf(v0, v1);
      #pragma unroll
      for (int off = 1; off < 16; off <<= 1)
        rm = fmaxf(rm, __shfl_xor(rm, off, 32));
      float mn = fmaxf(mrun[r], rm);
      float a  = (mrun[r] == -__builtin_inff())
                   ? ((mn == -__builtin_inff()) ? 1.f : 0.f)
                   : __expf(mrun[r] - mn);
      float p0 = (mn == -__builtin_inff()) ? 0.f : __expf(v0 - mn);
      float p1 = (mn == -__builtin_inff()) ? 0.f : __expf(v1 - mn);
      float rs = p0 + p1;
      #pragma unroll
      for (int off = 1; off < 16; off <<= 1)
        rs += __shfl_xor(rs, off, 32);
      lrun[r] = lrun[r] * a + rs;
      mrun[r] = mn;
      alpha[r] = a;
      Pls[w][(r + hi * 8) * 32 + mloc]      = (_Float16)p0;
      Pls[w][(r + hi * 8) * 32 + 16 + mloc] = (_Float16)p1;
    }
    #pragma unroll
    for (int r = 0; r < 8; ++r) {
      O0[r] *= alpha[r]; O1[r] *= alpha[r];
      O2[r] *= alpha[r]; O3[r] *= alpha[r];
    }

    // ---- O += P . V (P re-read in A layout; transposed V tile as B) ----
    asm volatile("s_wait_dscnt 0" ::: "memory");
    v16h pf  = load_a_frag(&Pls[w][0], mloc, 32, 0, hi);
    v16h vf0 = load_b_frag(Vtls,  0 + mloc, 32, 0, hi);
    v16h vf1 = load_b_frag(Vtls, 16 + mloc, 32, 0, hi);
    v16h vf2 = load_b_frag(Vtls, 32 + mloc, 32, 0, hi);
    v16h vf3 = load_b_frag(Vtls, 48 + mloc, 32, 0, hi);
    O0 = wmma_f16(pf, vf0, O0);
    O1 = wmma_f16(pf, vf1, O1);
    O2 = wmma_f16(pf, vf2, O2);
    O3 = wmma_f16(pf, vf3, O3);
    __syncthreads();
  }

  // ---- finalize: ctx = O / l, write (b, s, h) f16 ----
  #pragma unroll
  for (int r = 0; r < 8; ++r) {
    float inv = (lrun[r] > 0.f) ? (1.f / lrun[r]) : 0.f;
    int row = q0 + r + hi * 8;
    size_t base = ((size_t)bb * Sn + row) * Hn + head * HDn + mloc;
    cws[base + 0]  = (_Float16)(O0[r] * inv);
    cws[base + 16] = (_Float16)(O1[r] * inv);
    cws[base + 32] = (_Float16)(O2[r] * inv);
    cws[base + 48] = (_Float16)(O3[r] * inv);
  }
}

// ---------------------------------------------------------------------------
// Kernel 3: output projection. out = ctx(f16) . Wo^T + bo (fp32 out).
// Tile 128(M) x 64(N); ctx tile staged via async-to-LDS (already f16).
// ---------------------------------------------------------------------------
__global__ __launch_bounds__(256)
void out_proj_kernel(const _Float16* __restrict__ ctx,
                     const float* __restrict__ Wo, const float* __restrict__ bo,
                     float* __restrict__ out)
{
  __shared__ __align__(16) _Float16 Als[128 * 32];   // 8KB
  __shared__ __align__(16) _Float16 Bls[64 * 32];    // 4KB

  const int t  = threadIdx.x;
  const int m0 = blockIdx.x * 128;
  const int n0 = blockIdx.y * 64;
  const int lane = t & 31, w = t >> 5;
  const int hi = lane >> 4, mloc = lane & 15;

  const v8f vzero = {0.f, 0.f, 0.f, 0.f, 0.f, 0.f, 0.f, 0.f};
  v8f acc[4] = {vzero, vzero, vzero, vzero};

  for (int k0 = 0; k0 < Hn; k0 += 32) {
    if (k0 + 32 < Hn && t >= 128)
      __builtin_prefetch(Wo + (size_t)(n0 + (t & 63)) * Hn + (k0 + 32), 0, 3);
    // stage ctx tile 128x32 f16 via async DMA (2 b128 chunks per thread)
    #pragma unroll
    for (int i = 0; i < 2; ++i) {
      int c = t + i * 256;                 // 512 chunks of 8 halfs
      int row = c >> 2, dc = (c & 3) * 8;
      async_copy_b128(&Als[row * 32 + dc],
                      ctx + (size_t)(m0 + row) * Hn + k0 + dc);
    }
    // stage Wo tile 64(n) x 32(k) fp32 -> f16
    #pragma unroll
    for (int i = 0; i < 2; ++i) {
      int c = t + i * 256;
      int row = c >> 3, kc = (c & 7) * 4;
      float4 f = *(const float4*)(Wo + (size_t)(n0 + row) * Hn + k0 + kc);
      _Float16* dst = &Bls[row * 32 + kc];
      dst[0] = (_Float16)f.x; dst[1] = (_Float16)f.y;
      dst[2] = (_Float16)f.z; dst[3] = (_Float16)f.w;
    }
    async_wait0();
    __syncthreads();
    v16h a = load_a_frag(Als, w * 16 + mloc, 32, 0, hi);
    #pragma unroll
    for (int nb = 0; nb < 4; ++nb) {
      v16h b = load_b_frag(Bls, nb * 16 + mloc, 32, 0, hi);
      acc[nb] = wmma_f16(a, b, acc[nb]);
    }
    __syncthreads();
  }

  #pragma unroll
  for (int nb = 0; nb < 4; ++nb) {
    const int col = n0 + nb * 16 + mloc;
    const float bv_ = bo[col];
    #pragma unroll
    for (int r = 0; r < 8; ++r) {
      int gm = m0 + w * 16 + r + hi * 8;
      out[(size_t)gm * Hn + col] = acc[nb][r] + bv_;
    }
  }
}

// ---------------------------------------------------------------------------
extern "C" void kernel_launch(void* const* d_in, const int* in_sizes, int n_in,
                              void* d_out, int out_size, void* d_ws, size_t ws_size,
                              hipStream_t stream) {
  (void)in_sizes; (void)n_in; (void)out_size; (void)ws_size;

  const float* x    = (const float*)d_in[0];
  const int*   mask = (const int*)d_in[1];
  const float* Wq   = (const float*)d_in[2];
  const float* bq   = (const float*)d_in[3];
  const float* Wk   = (const float*)d_in[4];
  const float* bk   = (const float*)d_in[5];
  const float* Wv   = (const float*)d_in[6];
  const float* bv   = (const float*)d_in[7];
  const float* Wo   = (const float*)d_in[8];
  const float* bo   = (const float*)d_in[9];
  float* out = (float*)d_out;

  _Float16* qws = (_Float16*)d_ws;          // 16MB
  _Float16* kws = qws + QKV_ELEMS;          // 16MB
  _Float16* vws = kws + QKV_ELEMS;          // 16MB
  _Float16* cws = vws + QKV_ELEMS;          // 16MB (total 64MB workspace)

  dim3 blk(256);
  dim3 g1((Bn * Sn) / 128, (3 * Hn) / 64);  // 64 x 48
  qkv_proj_kernel<<<g1, blk, 0, stream>>>(x, Wq, bq, Wk, bk, Wv, bv,
                                          qws, kws, vws);

  dim3 g2(Sn / 128, NHn, Bn);               // 16 x 16 x 4
  flash_attn_kernel<<<g2, blk, 0, stream>>>(qws, kws, vws, mask, cws);

  dim3 g3((Bn * Sn) / 128, Hn / 64);        // 64 x 16
  out_proj_kernel<<<g3, blk, 0, stream>>>(cws, Wo, bo, out);
}
